// MoA_62869731279219
// MI455X (gfx1250) — matrix operand order repaired and checked by reading, not compiled, not run on hardware
//
#include <hip/hip_runtime.h>

// ---------------------------------------------------------------------------
// MoA (mixture-of-attention-heads) for MI455X / gfx1250, wave32 + bf16 WMMA.
// All matmul-shaped work runs on v_wmma_f32_16x16x32_bf16 (fp32 accumulate).
// ---------------------------------------------------------------------------

typedef __bf16 bf16_t;
typedef __attribute__((ext_vector_type(16))) __bf16 v16bf;
typedef __attribute__((ext_vector_type(8)))  __bf16 v8bf;
typedef __attribute__((ext_vector_type(8)))  float  v8f;

#define INPUT_SIZE 1024
#define OUTPUT_SIZE 1024
#define N_HEADS 8
#define N_EXPERTS 16
#define PROJ 128
#define BATCH 4
#define SEQ 1024
#define NTOK (BATCH * SEQ)          // 4096 tokens
#define ECOLS (N_EXPERTS * PROJ)    // 2048
#define KPOS_PAD 16
#define KPOS_ROWS (2048 + 2 * KPOS_PAD + 32)  // padded k_pos buffer rows (2112)

// ---- helpers ---------------------------------------------------------------

static __device__ __forceinline__ unsigned short f2bf_bits(float f) {
  union { float f; unsigned u; } c; c.f = f;
  unsigned r = c.u + 0x7FFFu + ((c.u >> 16) & 1u);   // round-to-nearest-even
  return (unsigned short)(r >> 16);
}
static __device__ __forceinline__ __bf16 f2bf(float f) {
  unsigned short h = f2bf_bits(f);
  __bf16 b; __builtin_memcpy(&b, &h, 2); return b;
}
static __device__ __forceinline__ v8f v8f_zero() {
  v8f z = {0.f,0.f,0.f,0.f,0.f,0.f,0.f,0.f}; return z;
}

// A-fragment (16x32, MxK): lane holds row (lane&15); k-chunks at
// k0 + 8*(lane>=16) and that +16 (ISA 7.12.2, 16-bit A layout).
static __device__ __forceinline__ v16bf load_afrag(const unsigned short* rowp, int koff) {
  v8bf lo = *reinterpret_cast<const v8bf*>(rowp + koff);
  v8bf hi = *reinterpret_cast<const v8bf*>(rowp + koff + 16);
  return __builtin_shufflevector(lo, hi, 0,1,2,3,4,5,6,7,8,9,10,11,12,13,14,15);
}
// B-fragment (32x16, KxN) from Bt[n][k] (row-major over K): lane holds column
// (lane&15); k-chunks at k0 + 8*(lane>=16) and that +16 (same order as A).
static __device__ __forceinline__ v16bf load_bfrag(const unsigned short* rowp, int koff) {
  v8bf lo = *reinterpret_cast<const v8bf*>(rowp + koff);
  v8bf hi = *reinterpret_cast<const v8bf*>(rowp + koff + 16);
  return __builtin_shufflevector(lo, hi, 0,1,2,3,4,5,6,7,8,9,10,11,12,13,14,15);
}
static __device__ __forceinline__ v8f wmma_bf16(v16bf a, v16bf b, v8f c) {
  return __builtin_amdgcn_wmma_f32_16x16x32_bf16(false, a, false, b, (short)0, c, false, false);
}

// ---- conversion / packing kernels -----------------------------------------

__global__ void conv_f32_bf16(const float* __restrict__ src,
                              unsigned short* __restrict__ dst, int n) {
  int i = blockIdx.x * blockDim.x + threadIdx.x;
  if (i < n) dst[i] = f2bf_bits(src[i]);
}

// src [batch][R][C] f32  ->  dst [batch][C][R] bf16
__global__ void transpose_conv(const float* __restrict__ src,
                               unsigned short* __restrict__ dst, int R, int C) {
  size_t bat = blockIdx.y;
  src += bat * (size_t)R * C;
  dst += bat * (size_t)R * C;
  int idx = blockIdx.x * blockDim.x + threadIdx.x;
  if (idx >= R * C) return;
  int c = idx / R, r = idx - c * R;
  dst[(size_t)c * R + r] = f2bf_bits(src[(size_t)r * C + c]);
}

// pemb[t][d], t in [0,2047] (row 2047 = zero pad), pos = t-1023.
__global__ void pemb_kernel(unsigned short* __restrict__ pemb) {
  int idx = blockIdx.x * blockDim.x + threadIdx.x;
  if (idx >= 2048 * INPUT_SIZE) return;
  int t = idx >> 10, d = idx & 1023;
  float val = 0.f;
  if (t < 2047) {
    float pos = (float)(t - 1023);
    float freq = __expf(-(float)(d & ~1) * (9.2103403719761836f / 1024.f));
    float ph = pos * freq;
    val = (d & 1) ? __cosf(ph) : __sinf(ph);
  }
  pemb[idx] = f2bf_bits(val);
}

// vt[p][token] = kv[token][128+p]  (bf16 transpose of V for probs@V WMMA)
__global__ void vtrans_kernel(const unsigned short* __restrict__ kv,
                              unsigned short* __restrict__ vt) {
  int idx = blockIdx.x * blockDim.x + threadIdx.x;
  if (idx >= PROJ * NTOK) return;
  int p = idx >> 12, tok = idx & (NTOK - 1);
  vt[idx] = kv[(size_t)tok * 256 + 128 + p];
}

// ---- selector + top-8 (one wave per token, fp32) ---------------------------

__global__ __launch_bounds__(32)
void sel_topk_kernel(const float* __restrict__ cs, const float* __restrict__ sel_dst,
                     float* __restrict__ selval, int* __restrict__ selidx) {
  int token = blockIdx.x;
  int lane = threadIdx.x;
  const float* row = cs + (size_t)token * INPUT_SIZE;
  float creg[32];
#pragma unroll
  for (int i = 0; i < 32; ++i) creg[i] = row[lane + 32 * i];
  float sv[N_EXPERTS];
#pragma unroll
  for (int e = 0; e < N_EXPERTS; ++e) {
    const float* w = sel_dst + (size_t)e * INPUT_SIZE;
    float a = 0.f;
#pragma unroll
    for (int i = 0; i < 32; ++i) a += creg[i] * w[lane + 32 * i];
#pragma unroll
    for (int m = 16; m >= 1; m >>= 1) a += __shfl_xor(a, m, 32);
    sv[e] = a;
  }
  bool taken[N_EXPERTS];
#pragma unroll
  for (int e = 0; e < N_EXPERTS; ++e) taken[e] = false;
  for (int k = 0; k < N_HEADS; ++k) {
    int bi = 0; float bv = -3.4e38f;
#pragma unroll
    for (int e = 0; e < N_EXPERTS; ++e)
      if (!taken[e] && sv[e] > bv) { bv = sv[e]; bi = e; }
    taken[bi] = true;
    if (lane == 0) {
      selval[(size_t)token * N_HEADS + k] = 1.f / (1.f + __expf(-bv));
      selidx[(size_t)token * N_HEADS + k] = bi;
    }
  }
}

// ---- generic bf16 WMMA GEMM:  C[MxN] = A[MxK] * Bt[NxK]^T ------------------
// One wave computes a 16x64 tile.  Out: bf16 or f32.

template <bool BF16OUT>
__global__ __launch_bounds__(256)
void gemm_bf16(const unsigned short* __restrict__ A,
               const unsigned short* __restrict__ Bt,
               void* __restrict__ Cv, int M, int N, int K) {
  int wave = (blockIdx.x * blockDim.x + threadIdx.x) >> 5;
  int lane = threadIdx.x & 31;
  int ntile64 = N >> 6;
  if (wave >= (M >> 4) * ntile64) return;
  int tm = wave / ntile64, tn = wave - tm * ntile64;
  int m0 = tm << 4, n0 = tn << 6;
  int half = lane >> 4, l16 = lane & 15;

  const unsigned short* arow = A + (size_t)(m0 + l16) * K;
  const unsigned short* brow0 = Bt + (size_t)(n0 + 0 * 16 + l16) * K;
  const unsigned short* brow1 = Bt + (size_t)(n0 + 1 * 16 + l16) * K;
  const unsigned short* brow2 = Bt + (size_t)(n0 + 2 * 16 + l16) * K;
  const unsigned short* brow3 = Bt + (size_t)(n0 + 3 * 16 + l16) * K;

  v8f acc0 = v8f_zero(), acc1 = v8f_zero(), acc2 = v8f_zero(), acc3 = v8f_zero();
  int aoff = half * 8, boff = half * 8;
  for (int k0 = 0; k0 < K; k0 += 32) {
    v16bf a = load_afrag(arow, k0 + aoff);
    acc0 = wmma_bf16(a, load_bfrag(brow0, k0 + boff), acc0);
    acc1 = wmma_bf16(a, load_bfrag(brow1, k0 + boff), acc1);
    acc2 = wmma_bf16(a, load_bfrag(brow2, k0 + boff), acc2);
    acc3 = wmma_bf16(a, load_bfrag(brow3, k0 + boff), acc3);
  }
#pragma unroll
  for (int r = 0; r < 8; ++r) {
    int m = m0 + r + 8 * half;
    size_t base = (size_t)m * N + n0 + l16;
    if (BF16OUT) {
      unsigned short* C = (unsigned short*)Cv;
      C[base +  0] = f2bf_bits(acc0[r]);
      C[base + 16] = f2bf_bits(acc1[r]);
      C[base + 32] = f2bf_bits(acc2[r]);
      C[base + 48] = f2bf_bits(acc3[r]);
    } else {
      float* C = (float*)Cv;
      C[base +  0] = acc0[r];
      C[base + 16] = acc1[r];
      C[base + 32] = acc2[r];
      C[base + 48] = acc3[r];
    }
  }
}

// ---- fused relative-position attention -------------------------------------
// One wave per (b, h, 16-row s-tile).  Scores kept in a 16x1024 f32 LDS strip.
// att[s,t] = scale * ( q[s].k[t] + q[s].k_pos[t-s+1023] ); softmax over t;
// res = att @ V ; output scattered (weighted by sigmoid gate) into dense z.

__global__ __launch_bounds__(32)
void attn_kernel(const unsigned short* __restrict__ qall,   // [NTOK][2048]
                 const unsigned short* __restrict__ kv,     // [NTOK][256]
                 const unsigned short* __restrict__ kpos,   // [KPOS_ROWS][128], +16 row offset
                 const unsigned short* __restrict__ vt,     // [128][NTOK]
                 const int* __restrict__ selidx,            // [NTOK][8]
                 const float* __restrict__ selval,          // [NTOK][8]
                 const float* __restrict__ scale,
                 unsigned short* __restrict__ z) {          // [NTOK][2048]
  __shared__ float strip[16 * 1024];

  int tile = blockIdx.x;
  int s0 = (tile & 63) << 4;
  int h  = (tile >> 6) & 7;
  int b  = tile >> 9;
  int lane = threadIdx.x;
  int half = lane >> 4, l16 = lane & 15;
  float scl = scale[0];

  // per-lane A row: q for token (b, s0+l16), expert chosen by head h
  int tokA = b * SEQ + s0 + l16;
  int eA = selidx[(size_t)tokA * N_HEADS + h];
  const unsigned short* qrow = qall + (size_t)tokA * ECOLS + eA * PROJ;
  v16bf qf[4];
#pragma unroll
  for (int kk = 0; kk < 4; ++kk) qf[kk] = load_afrag(qrow, kk * 32 + half * 8);

  int u0base = 1023 - s0;
  float vmax[8];
#pragma unroll
  for (int r = 0; r < 8; ++r) vmax[r] = -3.4e38f;

  // ---- phase 1: scores -> LDS strip ----
  for (int t0 = 0; t0 < SEQ; t0 += 16) {
    const unsigned short* krow  = kv   + (size_t)(b * SEQ + t0 + l16) * 256;
    int u0 = t0 + u0base;
    const unsigned short* p1row = kpos + (size_t)(u0 + l16 + KPOS_PAD) * PROJ;
    const unsigned short* p2row = kpos + (size_t)(u0 + l16) * PROJ;
    v8f acc = v8f_zero(), a1 = v8f_zero(), a2 = v8f_zero();
#pragma unroll
    for (int kk = 0; kk < 4; ++kk) {
      int bo = kk * 32 + half * 8;
      acc = wmma_bf16(qf[kk], load_bfrag(krow,  bo), acc);
      a1  = wmma_bf16(qf[kk], load_bfrag(p1row, bo), a1);
      a2  = wmma_bf16(qf[kk], load_bfrag(p2row, bo), a2);
    }
#pragma unroll
    for (int r = 0; r < 8; ++r) {
      int m = r + 8 * half;       // row of this lane's C element
      int c = l16 - m;            // needed k_pos column = u0 + c
      int src = (lane & 16) | (c & 15);
      float s1 = __shfl(a1[r], src, 32);
      float s2 = __shfl(a2[r], src, 32);
      float pv = (c >= 0) ? s1 : s2;
      float val = (acc[r] + pv) * scl;
      strip[m * 1024 + t0 + l16] = val;
      vmax[r] = fmaxf(vmax[r], val);
    }
  }
  __syncthreads();

  // ---- phase 2: softmax (row max / exp / sum), probs kept unnormalized ----
#pragma unroll
  for (int r = 0; r < 8; ++r) {
#pragma unroll
    for (int m = 8; m >= 1; m >>= 1) vmax[r] = fmaxf(vmax[r], __shfl_xor(vmax[r], m, 32));
  }
  float inv[8];
#pragma unroll
  for (int r = 0; r < 8; ++r) {
    int m = r + 8 * half;
    float* srow = strip + m * 1024;
    float mx = vmax[r], s = 0.f;
    for (int t = l16; t < SEQ; t += 16) {
      float ex = __expf(srow[t] - mx);
      srow[t] = ex;
      s += ex;
    }
#pragma unroll
    for (int mm = 8; mm >= 1; mm >>= 1) s += __shfl_xor(s, mm, 32);
    inv[r] = 1.0f / s;
  }
  __syncthreads();

  // ---- phase 3: res = probs @ V (WMMA), scatter to z ----
  v8f accv[8];
#pragma unroll
  for (int j = 0; j < 8; ++j) accv[j] = v8f_zero();

  for (int t0 = 0; t0 < SEQ; t0 += 32) {
    // A fragment (probs 16x32) from LDS with f32->bf16 conversion
    const float* srow = strip + l16 * 1024 + t0;
    v16bf af;
#pragma unroll
    for (int i = 0; i < 8; ++i) af[i] = f2bf(srow[half * 8 + i]);
#pragma unroll
    for (int i = 0; i < 8; ++i) af[8 + i] = f2bf(srow[16 + half * 8 + i]);
#pragma unroll
    for (int j = 0; j < 8; ++j) {
      const unsigned short* vrow = vt + (size_t)(j * 16 + l16) * NTOK + b * SEQ + t0;
      accv[j] = wmma_bf16(af, load_bfrag(vrow, half * 8), accv[j]);
    }
  }

#pragma unroll
  for (int r = 0; r < 8; ++r) {
    int m = r + 8 * half;
    int tok = b * SEQ + s0 + m;
    int e = selidx[(size_t)tok * N_HEADS + h];
    float g = selval[(size_t)tok * N_HEADS + h] * inv[r];
    unsigned short* zrow = z + (size_t)tok * ECOLS + e * PROJ + l16;
#pragma unroll
    for (int j = 0; j < 8; ++j) zrow[j * 16] = f2bf_bits(accv[j][r] * g);
  }
}

// ---------------------------------------------------------------------------

extern "C" void kernel_launch(void* const* d_in, const int* in_sizes, int n_in,
                              void* d_out, int out_size, void* d_ws, size_t ws_size,
                              hipStream_t stream) {
  (void)in_sizes; (void)n_in; (void)out_size; (void)ws_size;
  const float* curr_state = (const float*)d_in[0];
  const float* attend_to  = (const float*)d_in[1];
  const float* data_to_q  = (const float*)d_in[2];
  const float* data_to_kv = (const float*)d_in[3];
  const float* out_proj   = (const float*)d_in[4];
  const float* pos_to_pk  = (const float*)d_in[5];
  const float* sel_dst    = (const float*)d_in[6];
  const float* scale      = (const float*)d_in[7];
  float* out = (float*)d_out;

  char* ws = (char*)d_ws;
  size_t off = 0;
  auto alloc = [&](size_t bytes) { size_t r = off; off = (off + bytes + 255) & ~(size_t)255; return r; };
  unsigned short* cs_bf  = (unsigned short*)(ws + alloc((size_t)NTOK * 1024 * 2));
  unsigned short* at_bf  = (unsigned short*)(ws + alloc((size_t)NTOK * 1024 * 2));
  unsigned short* wq_t   = (unsigned short*)(ws + alloc((size_t)ECOLS * 1024 * 2));
  unsigned short* wkv_t  = (unsigned short*)(ws + alloc((size_t)256 * 1024 * 2));
  unsigned short* wpk    = (unsigned short*)(ws + alloc((size_t)128 * 1024 * 2));
  unsigned short* wout_t = (unsigned short*)(ws + alloc((size_t)1024 * ECOLS * 2));
  unsigned short* pemb   = (unsigned short*)(ws + alloc((size_t)2048 * 1024 * 2));
  float*          selval = (float*)(ws + alloc((size_t)NTOK * N_HEADS * 4));
  int*            selidx = (int*)(ws + alloc((size_t)NTOK * N_HEADS * 4));
  unsigned short* kv     = (unsigned short*)(ws + alloc((size_t)NTOK * 256 * 2));
  unsigned short* qall   = (unsigned short*)(ws + alloc((size_t)NTOK * ECOLS * 2));
  unsigned short* kposbf = (unsigned short*)(ws + alloc((size_t)KPOS_ROWS * PROJ * 2));
  unsigned short* vt     = (unsigned short*)(ws + alloc((size_t)PROJ * NTOK * 2));
  unsigned short* zbuf   = (unsigned short*)(ws + alloc((size_t)NTOK * ECOLS * 2));

  const int TB = 256;

  // 1) conversions / packing
  conv_f32_bf16<<<(NTOK * 1024 + TB - 1) / TB, TB, 0, stream>>>(curr_state, cs_bf, NTOK * 1024);
  conv_f32_bf16<<<(NTOK * 1024 + TB - 1) / TB, TB, 0, stream>>>(attend_to, at_bf, NTOK * 1024);
  conv_f32_bf16<<<(128 * 1024 + TB - 1) / TB, TB, 0, stream>>>(pos_to_pk, wpk, 128 * 1024);
  { dim3 g((1024 * 128 + TB - 1) / TB, N_EXPERTS);
    transpose_conv<<<g, TB, 0, stream>>>(data_to_q, wq_t, 1024, 128); }           // wq_t[e*128+p][k]
  { dim3 g((1024 * 256 + TB - 1) / TB, 1);
    transpose_conv<<<g, TB, 0, stream>>>(data_to_kv, wkv_t, 1024, 256); }         // wkv_t[n][i]
  { dim3 g((2048 * 1024 + TB - 1) / TB, 1);
    transpose_conv<<<g, TB, 0, stream>>>(out_proj, wout_t, 2048, 1024); }         // wout_t[o][e*128+p]
  pemb_kernel<<<(2048 * 1024 + TB - 1) / TB, TB, 0, stream>>>(pemb);
  hipMemsetAsync(kposbf, 0, (size_t)KPOS_ROWS * PROJ * 2, stream);
  hipMemsetAsync(zbuf, 0, (size_t)NTOK * ECOLS * 2, stream);

  // 2) selector + top-8 gating
  sel_topk_kernel<<<NTOK, 32, 0, stream>>>(curr_state, sel_dst, selval, selidx);

  // 3) WMMA GEMMs
  auto launch_gemm = [&](bool bfout, const unsigned short* A, const unsigned short* Bt,
                         void* C, int M, int N, int K) {
    int waves = (M / 16) * (N / 64);
    int blocks = (waves * 32 + TB - 1) / TB;
    if (bfout) gemm_bf16<true><<<blocks, TB, 0, stream>>>(A, Bt, C, M, N, K);
    else       gemm_bf16<false><<<blocks, TB, 0, stream>>>(A, Bt, C, M, N, K);
  };
  launch_gemm(true, at_bf, wkv_t, kv, NTOK, 256, 1024);                 // kv (k|v, unscaled)
  launch_gemm(true, cs_bf, wq_t, qall, NTOK, ECOLS, 1024);              // q for all experts
  launch_gemm(true, pemb, wpk, kposbf + (size_t)KPOS_PAD * PROJ,
              2048, PROJ, 1024);                                        // k_pos (padded rows)
  vtrans_kernel<<<(PROJ * NTOK + TB - 1) / TB, TB, 0, stream>>>(kv, vt);

  // 4) fused attention -> z (dense, gated expert scatter)
  attn_kernel<<<BATCH * N_HEADS * (SEQ / 16), 32, 0, stream>>>(
      qall, kv, kposbf, vt, selidx, selval, scale, zbuf);

  // 5) out = z @ out_proj (flattened over experts), fp32 result
  launch_gemm(false, zbuf, wout_t, out, NTOK, OUTPUT_SIZE, ECOLS);
}
